// MNISTnet_29094108463544
// MI455X (gfx1250) — compile-verified, matrix operand order checked by reading, not run on hardware
//
#include <hip/hip_runtime.h>

// ---------------- problem constants ----------------
#define T_STEPS 50
#define B       512
#define NE      400
#define NEP     416        // NE padded to multiple of 32 for STDP M-blocking
#define NIN     784
#define KP      800        // NIN padded to multiple of 32 for WMMA K loop
#define DT      0.5f

typedef __attribute__((ext_vector_type(16))) _Float16 v16h;
typedef __attribute__((ext_vector_type(8)))  float    v8f;

#define WMMA_F16(a, b, c) \
    __builtin_amdgcn_wmma_f32_16x16x32_f16(false, (a), false, (b), (short)0, (c), false, false)

// ---------------- init kernels ----------------

__global__ void k_copyW(const float* __restrict__ Win, float* __restrict__ W) {
    int i = blockIdx.x * blockDim.x + threadIdx.x;
    if (i < NE * NIN) W[i] = Win[i];
}

__global__ void k_zero_h(_Float16* __restrict__ p, int n) {
    int i = blockIdx.x * blockDim.x + threadIdx.x;
    if (i < n) p[i] = (_Float16)0.0f;
}

// convert x (T,B,NIN) fp32 -> xh (T,B,KP) f16 (zero padded) and xT (T,NIN,B) f16
__global__ void k_convert_x(const float* __restrict__ x,
                            _Float16* __restrict__ xh,
                            _Float16* __restrict__ xT) {
    long long idx = (long long)blockIdx.x * blockDim.x + threadIdx.x;
    long long total = (long long)T_STEPS * B * KP;
    if (idx >= total) return;
    int i  = (int)(idx % KP);
    long long tb = idx / KP;
    int b  = (int)(tb % B);
    int t  = (int)(tb / B);
    if (i < NIN) {
        float v = x[((long long)t * B + b) * NIN + i];
        xh[idx] = (_Float16)v;
        xT[((long long)t * NIN + i) * B + b] = (_Float16)v;
    } else {
        xh[idx] = (_Float16)0.0f;   // K padding
    }
}

__global__ void k_init_state(float* g_ine, float* g_ei, float* g_ie, float* Iie,
                             float* mem_e, float* rc_e, float* mem_i, float* rc_i,
                             float* theta, float* tp1, float* tp2) {
    int i = blockIdx.x * blockDim.x + threadIdx.x;
    if (i >= B * NE) return;
    g_ine[i] = 0.0f; g_ei[i] = 0.0f; g_ie[i] = 0.0f; Iie[i] = 0.0f;
    mem_e[i] = -65.0f; rc_e[i] = 0.0f;
    mem_i[i] = -60.0f; rc_i[i] = 0.0f;
    theta[i] = 20.0f; tp1[i] = 0.0f; tp2[i] = 0.0f;
}

__global__ void k_init_trpre(float* __restrict__ tr32) {
    int i = blockIdx.x * blockDim.x + threadIdx.x;
    if (i < NIN * B) tr32[i] = 0.0f;
}

// ---------------- per-step kernels ----------------

// normweight: scale each W row to sum 78.4; refresh f16 copy (padded cols stay 0)
__global__ void k_normalize(float* __restrict__ W, _Float16* __restrict__ Wh) {
    __shared__ float red[256];
    int row = blockIdx.x;
    float s = 0.0f;
    for (int i = threadIdx.x; i < NIN; i += 256) s += W[row * NIN + i];
    red[threadIdx.x] = s;
    __syncthreads();
    for (int off = 128; off > 0; off >>= 1) {
        if (threadIdx.x < off) red[threadIdx.x] += red[threadIdx.x + off];
        __syncthreads();
    }
    float scale = 78.4f / red[0];
    for (int i = threadIdx.x; i < NIN; i += 256) {
        float w = W[row * NIN + i] * scale;
        W[row * NIN + i]  = w;
        Wh[row * KP + i]  = (_Float16)w;
    }
}

// dg_ine = xt @ W^T : M=B(512), N=NE(400), K=KP(800)
// 4x M register blocking + manual double-buffered pipeline:
// iteration k's WMMAs consume operands loaded in iteration k-1,
// so the in-loop s_wait_loadcnt does not stall on the fresh loads.
__global__ void k_gemm_dgine(const _Float16* __restrict__ xh,   // (B, KP)
                             const _Float16* __restrict__ Wh,   // (NE, KP)
                             float* __restrict__ dg) {          // (B, NE)
    int wave = (blockIdx.x * blockDim.x + threadIdx.x) >> 5;
    int lane = threadIdx.x & 31;
    if (wave >= 8 * 25) return;                  // wave-uniform guard
    int tn  = wave % 25;
    int tmb = wave / 25;                         // 64-row M block
    int m0   = tmb * 64 + (lane & 15);
    int bcol = tn * 16 + (lane & 15);
    int kh   = (lane >> 4) * 16;
    const _Float16* ap = xh + (long long)m0 * KP + kh;
    const _Float16* bp = Wh + (long long)bcol * KP + kh;

    v8f acc0 = {}, acc1 = {}, acc2 = {}, acc3 = {};
    // prologue: load K-slice 0
    v16h bv = *(const v16h*)(bp);
    v16h a0 = *(const v16h*)(ap);
    v16h a1 = *(const v16h*)(ap + 16 * KP);
    v16h a2 = *(const v16h*)(ap + 32 * KP);
    v16h a3 = *(const v16h*)(ap + 48 * KP);
    for (int k = 32; k < KP; k += 32) {
        // issue next K-slice loads into fresh registers
        v16h bn = *(const v16h*)(bp + k);
        v16h n0 = *(const v16h*)(ap + k);
        v16h n1 = *(const v16h*)(ap + 16 * KP + k);
        v16h n2 = *(const v16h*)(ap + 32 * KP + k);
        v16h n3 = *(const v16h*)(ap + 48 * KP + k);
        // consume previous slice (already resident)
        acc0 = WMMA_F16(a0, bv, acc0);
        acc1 = WMMA_F16(a1, bv, acc1);
        acc2 = WMMA_F16(a2, bv, acc2);
        acc3 = WMMA_F16(a3, bv, acc3);
        bv = bn; a0 = n0; a1 = n1; a2 = n2; a3 = n3;
    }
    // epilogue: last slice
    acc0 = WMMA_F16(a0, bv, acc0);
    acc1 = WMMA_F16(a1, bv, acc1);
    acc2 = WMMA_F16(a2, bv, acc2);
    acc3 = WMMA_F16(a3, bv, acc3);

    int rbase = (lane < 16) ? 0 : 8;             // C layout: VGPR r -> M=r / r+8
    int col   = tn * 16 + (lane & 15);
    #pragma unroll
    for (int r = 0; r < 8; ++r) {
        int mb = tmb * 64 + rbase + r;
        dg[(long long)(mb +  0) * NE + col] = acc0[r];
        dg[(long long)(mb + 16) * NE + col] = acc1[r];
        dg[(long long)(mb + 32) * NE + col] = acc2[r];
        dg[(long long)(mb + 48) * NE + col] = acc3[r];
    }
}

// fused excitatory LIF + inhibitory LIF + per-batch spike reduction.
// one block per batch element; spike_e never leaves registers,
// spike_i reduced in LDS (W_ie = 17*(ones - I) analytic trick).
__global__ void k_lif(const float* __restrict__ dg_ine,
                      float* g_ine, float* Iie,
                      float* mem_e, float* rc_e, float* theta,
                      float* tp1, float* tp2,
                      float* g_ei, float* mem_i, float* rc_i, float* g_ie,
                      float* __restrict__ out_t,           // d_out + t*B*NE
                      _Float16* __restrict__ ret1T,        // (NEP,B) f16
                      _Float16* __restrict__ spret2T) {    // (NEP,B) f16
    __shared__ float sspike[NE];
    __shared__ float red[128];
    int b   = blockIdx.x;
    int tid = threadIdx.x;
    float mem_loc[4];
    float part = 0.0f;
    #pragma unroll
    for (int it = 0; it < 4; ++it) {
        int e = tid + it * 128;
        if (e < NE) {
            int i = b * NE + e;
            // ---- excitatory LIF ----
            float g  = g_ine[i];
            float m  = mem_e[i];
            float I  = g * (0.0f - m) + Iie[i];          // E_EE = 0
            g += (dg_ine[i] - g) * (DT / 1.0f);          // TAU_SYN_E = 1
            float rc = rc_e[i];
            bool active = (rc <= 0.0f);
            if (active) m += (-65.0f - m + I) * (DT / 100.0f);
            rc = fmaxf(rc - DT, 0.0f);
            float th  = theta[i];
            float thr = -52.0f - 20.0f + th;             // THRESH_E - OFFSET + theta
            float spike = (active && (m > thr)) ? 1.0f : 0.0f;
            if (spike > 0.0f) { m = -65.0f; rc = 5.0f; }
            th = th * (1.0f - DT / 1.0e7f) + 0.05f * spike;

            float r1 = tp1[i] * (1.0f - DT / 20.0f);
            float r2 = tp2[i] * (1.0f - DT / 40.0f);
            tp1[i] = (spike > 0.9f) ? 1.0f : r1;
            tp2[i] = (spike > 0.9f) ? 1.0f : r2;

            g_ine[i] = g; mem_e[i] = m; rc_e[i] = rc; theta[i] = th;
            mem_loc[it] = m;
            out_t[i] = spike;
            ret1T[(long long)e * B + b]   = (_Float16)r1;
            spret2T[(long long)e * B + b] = (_Float16)(spike * r2);

            // ---- inhibitory LIF (dg_ei = 10.4 * spike_e, W_ei = 10.4*I) ----
            float mi   = mem_i[i];
            float I_ei = g_ei[i] * (0.0f - mi);          // E_EI = 0
            g_ei[i] += (10.4f * spike - g_ei[i]) * (DT / 1.0f);
            float rci = rc_i[i];
            bool acti = (rci <= 0.0f);
            if (acti) mi += (-60.0f - mi + I_ei) * (DT / 10.0f);
            rci = fmaxf(rci - DT, 0.0f);
            float spi = (acti && (mi > -40.0f)) ? 1.0f : 0.0f;
            if (spi > 0.0f) { mi = -45.0f; rci = 2.0f; }
            mem_i[i] = mi; rc_i[i] = rci;
            sspike[e] = spi;
            part += spi;
        } else {
            mem_loc[it] = 0.0f;
        }
    }
    red[tid] = part;
    __syncthreads();
    for (int off = 64; off > 0; off >>= 1) {
        if (tid < off) red[tid] += red[tid + off];
        __syncthreads();
    }
    float S = red[0];
    #pragma unroll
    for (int it = 0; it < 4; ++it) {
        int e = tid + it * 128;
        if (e < NE) {
            int i = b * NE + e;
            float dg_ie = 17.0f * (S - sspike[e]);       // W_ie = 17*(ones - I)
            float g = g_ie[i];
            Iie[i]  = g * (-100.0f - mem_loc[it]);       // E_IE = -100, post-update mem_e
            g_ie[i] = g + (dg_ie - g) * (DT / 2.0f);     // TAU_SYN_I = 2
        }
    }
}

// pre-synaptic trace update in transposed (NIN,B) layout
__global__ void k_trpre(const _Float16* __restrict__ xTt,   // (NIN,B) f16, step t
                        float* __restrict__ tr32,
                        _Float16* __restrict__ trT,
                        int is_first) {
    int i = blockIdx.x * blockDim.x + threadIdx.x;
    if (i >= NIN * B) return;
    float xv = (float)xTt[i];
    float tr;
    if (is_first) tr = xv;
    else          tr = (xv > 0.9f) ? 1.0f : tr32[i] * (1.0f - DT / 20.0f);
    tr32[i] = tr;
    trT[i]  = (_Float16)tr;
}

// STDP: dw[e,i] = NU_PRE * sum_b ret1T[e,b]*xT[i,b] + NU_POST * sum_b spret2T[e,b]*trT[i,b]
// 2x M register blocking (shared B-operand per chain) + double-buffered pipeline.
// waves = (NEP/32) x (NIN/16) = 13 x 49 = 637; rows >= NE are zero padding.
__global__ void k_stdp(const _Float16* __restrict__ ret1T,    // (NEP,B)
                       const _Float16* __restrict__ spret2T,  // (NEP,B)
                       const _Float16* __restrict__ xTt,      // (NIN,B)
                       const _Float16* __restrict__ trT,      // (NIN,B)
                       float* __restrict__ W) {               // (NE,NIN)
    int wave = (blockIdx.x * blockDim.x + threadIdx.x) >> 5;
    int lane = threadIdx.x & 31;
    if (wave >= 13 * 49) return;                 // wave-uniform guard
    int ti = wave % 49;
    int tp = wave / 49;                          // 32-row M block over e
    int e0   = tp * 32 + (lane & 15);
    int icol = ti * 16 + (lane & 15);
    int kh   = (lane >> 4) * 16;
    const _Float16* a1 = ret1T   + (long long)e0 * B + kh;
    const _Float16* a2 = spret2T + (long long)e0 * B + kh;
    const _Float16* b1 = xTt     + (long long)icol * B + kh;
    const _Float16* b2 = trT     + (long long)icol * B + kh;

    v8f acc1lo = {}, acc1hi = {}, acc2lo = {}, acc2hi = {};
    // prologue: K-slice 0
    v16h vb1  = *(const v16h*)(b1);
    v16h vb2  = *(const v16h*)(b2);
    v16h a1lo = *(const v16h*)(a1);
    v16h a1hi = *(const v16h*)(a1 + 16 * B);
    v16h a2lo = *(const v16h*)(a2);
    v16h a2hi = *(const v16h*)(a2 + 16 * B);
    for (int k = 32; k < B; k += 32) {
        v16h nb1  = *(const v16h*)(b1 + k);
        v16h nb2  = *(const v16h*)(b2 + k);
        v16h n1lo = *(const v16h*)(a1 + k);
        v16h n1hi = *(const v16h*)(a1 + 16 * B + k);
        v16h n2lo = *(const v16h*)(a2 + k);
        v16h n2hi = *(const v16h*)(a2 + 16 * B + k);
        acc1lo = WMMA_F16(a1lo, vb1, acc1lo);
        acc1hi = WMMA_F16(a1hi, vb1, acc1hi);
        acc2lo = WMMA_F16(a2lo, vb2, acc2lo);
        acc2hi = WMMA_F16(a2hi, vb2, acc2hi);
        vb1 = nb1; vb2 = nb2;
        a1lo = n1lo; a1hi = n1hi; a2lo = n2lo; a2hi = n2hi;
    }
    acc1lo = WMMA_F16(a1lo, vb1, acc1lo);
    acc1hi = WMMA_F16(a1hi, vb1, acc1hi);
    acc2lo = WMMA_F16(a2lo, vb2, acc2lo);
    acc2hi = WMMA_F16(a2hi, vb2, acc2hi);

    int rbase = (lane < 16) ? 0 : 8;
    int col   = ti * 16 + (lane & 15);
    #pragma unroll
    for (int r = 0; r < 8; ++r) {
        int elo = tp * 32 + rbase + r;
        int ehi = elo + 16;
        if (elo < NE) {
            long long wi = (long long)elo * NIN + col;
            float w = W[wi] + 1.0e-4f * acc1lo[r] + 0.01f * acc2lo[r];
            W[wi] = fminf(fmaxf(w, 0.0f), 1.0f);
        }
        if (ehi < NE) {
            long long wi = (long long)ehi * NIN + col;
            float w = W[wi] + 1.0e-4f * acc1hi[r] + 0.01f * acc2hi[r];
            W[wi] = fminf(fmaxf(w, 0.0f), 1.0f);
        }
    }
}

// ---------------- host side ----------------

static inline size_t alignUp256(size_t x) { return (x + 255) & ~(size_t)255; }

extern "C" void kernel_launch(void* const* d_in, const int* in_sizes, int n_in,
                              void* d_out, int out_size, void* d_ws, size_t ws_size,
                              hipStream_t stream) {
    (void)in_sizes; (void)n_in; (void)out_size; (void)ws_size;
    const float* x   = (const float*)d_in[0];    // (T,B,NIN)
    const float* Win = (const float*)d_in[1];    // (NE,NIN)
    float* out = (float*)d_out;                  // (T,B,NE)

    char* ws = (char*)d_ws;
    size_t off = 0;
    auto carve = [&](size_t bytes) -> void* {
        void* p = ws + off; off = alignUp256(off + bytes); return p;
    };

    float*    W       = (float*)   carve((size_t)NE * NIN * 4);
    _Float16* Wh      = (_Float16*)carve((size_t)NE * KP * 2);
    _Float16* xh      = (_Float16*)carve((size_t)T_STEPS * B * KP * 2);
    _Float16* xT      = (_Float16*)carve((size_t)T_STEPS * NIN * B * 2);
    float*    dgine   = (float*)   carve((size_t)B * NE * 4);
    float*    g_ine   = (float*)   carve((size_t)B * NE * 4);
    float*    g_ei    = (float*)   carve((size_t)B * NE * 4);
    float*    g_ie    = (float*)   carve((size_t)B * NE * 4);
    float*    Iie     = (float*)   carve((size_t)B * NE * 4);
    float*    mem_e   = (float*)   carve((size_t)B * NE * 4);
    float*    rc_e    = (float*)   carve((size_t)B * NE * 4);
    float*    mem_i   = (float*)   carve((size_t)B * NE * 4);
    float*    rc_i    = (float*)   carve((size_t)B * NE * 4);
    float*    theta   = (float*)   carve((size_t)B * NE * 4);
    float*    tp1     = (float*)   carve((size_t)B * NE * 4);
    float*    tp2     = (float*)   carve((size_t)B * NE * 4);
    float*    tr32    = (float*)   carve((size_t)NIN * B * 4);
    _Float16* trT     = (_Float16*)carve((size_t)NIN * B * 2);
    _Float16* ret1T   = (_Float16*)carve((size_t)NEP * B * 2);
    _Float16* spret2T = (_Float16*)carve((size_t)NEP * B * 2);

    // ---- per-launch init (deterministic across replays) ----
    {
        int n = NE * NIN;
        k_copyW<<<(n + 255) / 256, 256, 0, stream>>>(Win, W);
    }
    {
        int n = NE * KP;
        k_zero_h<<<(n + 255) / 256, 256, 0, stream>>>(Wh, n);       // K padding
    }
    {
        int n = NEP * B;
        k_zero_h<<<(n + 255) / 256, 256, 0, stream>>>(ret1T, n);    // M padding rows
        k_zero_h<<<(n + 255) / 256, 256, 0, stream>>>(spret2T, n);
    }
    {
        long long n = (long long)T_STEPS * B * KP;
        k_convert_x<<<(unsigned)((n + 255) / 256), 256, 0, stream>>>(x, xh, xT);
    }
    {
        int n = B * NE;
        k_init_state<<<(n + 255) / 256, 256, 0, stream>>>(
            g_ine, g_ei, g_ie, Iie, mem_e, rc_e, mem_i, rc_i, theta, tp1, tp2);
    }
    {
        int n = NIN * B;
        k_init_trpre<<<(n + 255) / 256, 256, 0, stream>>>(tr32);
    }

    // ---- 50 sequential SNN steps (5 kernels/step) ----
    for (int t = 0; t < T_STEPS; ++t) {
        k_normalize<<<NE, 256, 0, stream>>>(W, Wh);

        // 200 waves, 8 waves/block
        k_gemm_dgine<<<25, 256, 0, stream>>>(
            xh + (size_t)t * B * KP, Wh, dgine);

        k_lif<<<B, 128, 0, stream>>>(
            dgine, g_ine, Iie, mem_e, rc_e, theta, tp1, tp2,
            g_ei, mem_i, rc_i, g_ie,
            out + (size_t)t * B * NE, ret1T, spret2T);

        k_trpre<<<(NIN * B + 255) / 256, 256, 0, stream>>>(
            xT + (size_t)t * NIN * B, tr32, trT, (t == 0) ? 1 : 0);

        // 637 waves, 8 waves/block -> 80 blocks (partial last)
        k_stdp<<<80, 256, 0, stream>>>(
            ret1T, spret2T, xT + (size_t)t * NIN * B, trT, W);
    }
}
